// FractalAttentionalResonance_19816979104520
// MI455X (gfx1250) — compile-verified, matrix operand order checked by reading, not run on hardware
//
#include <hip/hip_runtime.h>

#define B_  4
#define S_  2048
#define D_  512
#define H_  8
#define HD_ 64
#define BS_ (B_*S_)   // 8192 rows

typedef __attribute__((ext_vector_type(16))) __bf16 v16bf;
typedef __attribute__((ext_vector_type(8)))  float  v8f;
typedef __attribute__((ext_vector_type(4)))  unsigned int u32x4;
typedef __attribute__((ext_vector_type(4)))  int i32x4;
typedef __attribute__((ext_vector_type(8)))  int i32x8;
typedef int gv4i __attribute__((vector_size(16)));   // matches async builtin pointee

union Frag { v16bf v; u32x4 q[2]; unsigned int u[8]; unsigned short h[16]; };

// ---------- CDNA5 async global->LDS copy (ASYNCcnt) with safe fallback ------
#if defined(__has_builtin)
#if __has_builtin(__builtin_amdgcn_global_load_async_to_lds_b128)
#define HAVE_ASYNC 1
#endif
#if __has_builtin(__builtin_amdgcn_tensor_load_to_lds)
#define HAVE_TDM 1
#endif
#endif
#ifndef HAVE_ASYNC
#define HAVE_ASYNC 0
#endif
#ifndef HAVE_TDM
#define HAVE_TDM 0
#endif

__device__ __forceinline__ void cp16(const void* g, void* l) {
#if HAVE_ASYNC
  __builtin_amdgcn_global_load_async_to_lds_b128(
      (__attribute__((address_space(1))) gv4i*)(g),
      (__attribute__((address_space(3))) gv4i*)(l), 0, 0);
#else
  *(u32x4*)l = *(const u32x4*)g;
#endif
}
__device__ __forceinline__ void cp_commit() {
#if HAVE_ASYNC
#if __has_builtin(__builtin_amdgcn_s_wait_asynccnt)
  __builtin_amdgcn_s_wait_asynccnt(0);
#else
  asm volatile("s_wait_asynccnt 0" ::: "memory");
#endif
#endif
}

#if HAVE_TDM
// 2D bf16 tile DMA via Tensor Data Mover: 32x64-elem tile, row stride 64 elems,
// destination LDS gets 16B padding after every 128B (-> stride-72 rows).
__device__ __forceinline__ void tdm_load_k_tile(const void* gsrc, unsigned lds_off) {
  unsigned long long ga = (unsigned long long)(size_t)gsrc;
  u32x4 g0;
  g0[0] = 1u;                                          // count=1, user D#
  g0[1] = lds_off;                                     // lds_addr (bytes)
  g0[2] = (unsigned)(ga & 0xffffffffu);                // global_addr[31:0]
  g0[3] = (unsigned)((ga >> 32) & 0x1ffffffu) | (2u << 30);  // addr[56:32], type=2
  i32x8 g1;
  g1[0] = (1 << 16)      // data_size = 2 bytes
        | (1 << 20)      // pad_enable
        | (4 << 22)      // pad_interval: every 32 dwords (128B)
        | (3 << 25);     // pad_amount: 4 dwords (16B)
  g1[1] = (int)(64u << 16);   // tensor_dim0 = 64 elems (bits 79:48, low half)
  g1[2] = (int)(32u << 16);   // tensor_dim1 = 32 rows (bits 111:80, low half)
  g1[3] = (int)(64u << 16);   // tile_dim0 = 64 (bits 127:112)
  g1[4] = 32;                 // tile_dim1 = 32 (bits 143:128)
  g1[5] = 64;                 // tensor_dim0_stride = 64 elems (bits 207:160)
  g1[6] = 0;
  g1[7] = 0;
  i32x4 g2 = {0, 0, 0, 0};
  i32x4 g3 = {0, 0, 0, 0};
  i32x8 g4 = {0, 0, 0, 0, 0, 0, 0, 0};
  __builtin_amdgcn_tensor_load_to_lds(g0, g1, g2, g3, g4, 0);
}
__device__ __forceinline__ void tdm_commit() {
#if __has_builtin(__builtin_amdgcn_s_wait_tensorcnt)
  __builtin_amdgcn_s_wait_tensorcnt((short)0);
#else
  asm volatile("s_wait_tensorcnt 0" ::: "memory");
#endif
}
#endif

__device__ __forceinline__ unsigned short f2bf(float f) {
  unsigned int u = __builtin_bit_cast(unsigned int, f);
  u += 0x7fffu + ((u >> 16) & 1u);
  return (unsigned short)(u >> 16);
}
__device__ __forceinline__ float bf2f(unsigned short s) {
  unsigned int u = ((unsigned int)s) << 16;
  return __builtin_bit_cast(float, u);
}

__device__ __forceinline__ v8f wmma_bf16(const Frag& a, const Frag& b, v8f c) {
  return __builtin_amdgcn_wmma_f32_16x16x32_bf16(false, a.v, false, b.v,
                                                 (short)0, c, false, false);
}

// A-fragment 16x32 bf16 from row-major LDS; strideU*2 and colBase*2 must be
// multiples of 16 bytes -> exactly 2x ds_load_b128 per fragment.
__device__ __forceinline__ Frag ldsA16(const unsigned short* p, int strideU,
                                       int rowBase, int colBase) {
  Frag f;
  int lane = threadIdx.x & 31, half = lane >> 4, r = lane & 15;
  const unsigned short* q = p + (rowBase + r) * strideU + colBase + half * 8;
  f.q[0] = *(const u32x4*)(q);
  f.q[1] = *(const u32x4*)(q + 16);
  return f;
}

// Swizzled B tile: dword (n,p) = {B[2p][n], B[2p+1][n]} at n*16 + swz-chunk.
__device__ __forceinline__ int bswz(int n, int p) {
  return n * 16 + ((((p >> 2) ^ ((n >> 2) & 3)) << 2) | (p & 3));
}
// B-fragment 32x16 from swizzled dword tile -> 2x ds_load_b128.
__device__ __forceinline__ Frag ldsBsw(const unsigned int* base, int nBase) {
  Frag f;
  int lane = threadIdx.x & 31, half = lane >> 4, c = lane & 15;
  int n = nBase + c, sw = (n >> 2) & 3;
  f.q[0] = *(const u32x4*)(base + n * 16 + ((half ^ sw) << 2));
  f.q[1] = *(const u32x4*)(base + n * 16 + (((half + 2) ^ sw) << 2));
  return f;
}

// ---------------- kernel 1: x + tf_emb[ids] -> bf16 ----------------
__global__ void k_add_emb(const float* __restrict__ x, const int* __restrict__ ids,
                          const float* __restrict__ tf, unsigned short* __restrict__ xb) {
  int i4 = blockIdx.x * 256 + threadIdx.x;      // over B*S*D/4
  int idx = i4 * 4;
  int row = idx >> 9, d = idx & (D_ - 1);
  int id = ids[row];
  const float* xp = x + idx;
  const float* tp = tf + id * D_ + d;
  unsigned int lo = f2bf(xp[0] + tp[0]) | ((unsigned)f2bf(xp[1] + tp[1]) << 16);
  unsigned int hi = f2bf(xp[2] + tp[2]) | ((unsigned)f2bf(xp[3] + tp[3]) << 16);
  *(unsigned int*)&xb[idx]     = lo;
  *(unsigned int*)&xb[idx + 2] = hi;
}

// ---------------- kernel 2: gc = mean over S ----------------
__global__ void k_gc(const unsigned short* __restrict__ xb, float* __restrict__ gc) {
  int idx = blockIdx.x * 256 + threadIdx.x;     // over B*D
  int b = idx >> 9, d = idx & (D_ - 1);
  const unsigned short* p = xb + (size_t)b * S_ * D_ + d;
  float s = 0.f;
  for (int t = 0; t < S_; ++t) s += bf2f(p[(size_t)t * D_]);
  gc[idx] = s * (1.0f / S_);
}

// ---------------- kernel 3: fractal bias MLP ----------------
__global__ void k_fb(const float* __restrict__ gc, const float* __restrict__ Wc1,
                     const float* __restrict__ bc1, const float* __restrict__ Wc2,
                     const float* __restrict__ bc2, const float* __restrict__ rg,
                     float* __restrict__ fb) {
  __shared__ float hid[D_ / 2];
  int b = blockIdx.x, t = threadIdx.x;
  const float* g = gc + b * D_;
  float a = bc1[t];
  for (int k = 0; k < D_; ++k) a += g[k] * Wc1[k * (D_ / 2) + t];
  hid[t] = fmaxf(a, 0.f);
  __syncthreads();
  if (t < H_) {
    float s = bc2[t];
    for (int k = 0; k < D_ / 2; ++k) s += hid[k] * Wc2[k * H_ + t];
    float gate = 1.f / (1.f + __expf(-rg[t]));
    fb[b * H_ + t] = tanhf(s) * gate;
  }
}

// ---------------- kernel 4: QKV projection GEMM (bf16 WMMA) ----------------
// LAYOUT 1: row-major (B,H,S,HD) bf16 (Q,K); LAYOUT 2: V pair-swizzled tiles.
template <int LAYOUT>
__global__ __launch_bounds__(256) void k_gemm_proj(
    const unsigned short* __restrict__ A, const float* __restrict__ W,
    const float* __restrict__ bias, unsigned short* __restrict__ Out) {
  __shared__ __attribute__((aligned(16))) unsigned short As[128 * 40];
  __shared__ __attribute__((aligned(16))) unsigned int   Bsw[64 * 16];
  int tid = threadIdx.x;
  int w = tid >> 5, wm = w >> 1, wn = w & 1;
  int lane = tid & 31, half = lane >> 4, r = lane & 15;
  int mBase = blockIdx.x * 128, nBase = blockIdx.y * 64;

  v8f z;
#pragma unroll
  for (int i = 0; i < 8; ++i) z[i] = 0.f;
  v8f c00 = z, c01 = z, c10 = z, c11 = z;

  for (int kb = 0; kb < D_; kb += 32) {
    __syncthreads();
#pragma unroll
    for (int i = 0; i < 2; ++i) {                // A tile: 512 x 16B chunks
      int cidx = i * 256 + tid;
      int row = cidx >> 2, seg = cidx & 3;
      cp16(&A[(size_t)(mBase + row) * D_ + kb + seg * 8], &As[row * 40 + seg * 8]);
    }
#pragma unroll
    for (int i = 0; i < 4; ++i) {                // B tile 32x64 f32 -> swizzled bf16
      int u = i * 256 + tid;
      int n = u & 63, p = u >> 6;
      float lo = W[(size_t)(kb + 2 * p)     * D_ + nBase + n];
      float hi = W[(size_t)(kb + 2 * p + 1) * D_ + nBase + n];
      Bsw[bswz(n, p)] = (unsigned)f2bf(lo) | ((unsigned)f2bf(hi) << 16);
    }
    cp_commit();
    __syncthreads();

    Frag a0 = ldsA16(As, 40, wm * 32,      0);
    Frag a1 = ldsA16(As, 40, wm * 32 + 16, 0);
    Frag b0 = ldsBsw(Bsw, wn * 32);
    Frag b1 = ldsBsw(Bsw, wn * 32 + 16);
    c00 = wmma_bf16(a0, b0, c00);
    c01 = wmma_bf16(a0, b1, c01);
    c10 = wmma_bf16(a1, b0, c10);
    c11 = wmma_bf16(a1, b1, c11);
  }

  v8f cs[2][2] = {{c00, c01}, {c10, c11}};
#pragma unroll
  for (int sub = 0; sub < 2; ++sub)
#pragma unroll
    for (int nt = 0; nt < 2; ++nt)
#pragma unroll
      for (int j = 0; j < 8; ++j) {
        int m = mBase + wm * 32 + sub * 16 + j + half * 8;
        int n = nBase + wn * 32 + nt * 16 + r;
        float v = cs[sub][nt][j] + bias[n];
        int b = m >> 11, s = m & (S_ - 1);
        int hh = n >> 6, hd = n & 63;
        int bh = b * H_ + hh;
        if (LAYOUT == 1) {
          Out[(((size_t)bh) * S_ + s) * HD_ + hd] = f2bf(v);
        } else {                                 // V: pair-swizzled 32-key tiles
          int T = s >> 5, w32 = s & 31, p = w32 >> 1, par = w32 & 1;
          size_t dw = ((size_t)(bh * (S_ / 32) + T)) * 1024 + bswz(hd, p);
          Out[dw * 2 + par] = f2bf(v);
        }
      }
}

// ---------------- kernel 5: flash attention (bf16 WMMA, online softmax) ------
__global__ __launch_bounds__(256) void k_flash(
    const unsigned short* __restrict__ Q, const unsigned short* __restrict__ K,
    const unsigned short* __restrict__ Vsw, const int* __restrict__ mask,
    const float* __restrict__ fb, unsigned short* __restrict__ Obuf) {
  __shared__ __attribute__((aligned(16))) unsigned short Kt[32 * 72];
  __shared__ __attribute__((aligned(16))) unsigned int   Vt[1024];
  __shared__ __attribute__((aligned(16))) unsigned short Ps[8 * 16 * 40];

  int tid = threadIdx.x, w = tid >> 5, lane = tid & 31;
  int half = lane >> 4, r = lane & 15;
  int bh = blockIdx.y, b = bh >> 3, h = bh & 7;
  int qBase = blockIdx.x * 128 + w * 16;
  const size_t kvBase = (size_t)bh * S_ * HD_;
  const unsigned int* vtiles = (const unsigned int*)Vsw + (size_t)bh * (S_ / 32) * 1024;

  Frag aq0, aq1;                                 // Q strip: 4x global b128/lane
  {
    const unsigned short* p = Q + ((size_t)bh * S_ + qBase + r) * HD_ + half * 8;
    aq0.q[0] = *(const u32x4*)(p);
    aq0.q[1] = *(const u32x4*)(p + 16);
    aq1.q[0] = *(const u32x4*)(p + 32);
    aq1.q[1] = *(const u32x4*)(p + 48);
  }
  float fbias = fb[b * H_ + h];
  const float scale = 0.125f;
  float m[8], l[8];
#pragma unroll
  for (int j = 0; j < 8; ++j) { m[j] = -1e30f; l[j] = 0.f; }
  v8f z;
#pragma unroll
  for (int i = 0; i < 8; ++i) z[i] = 0.f;
  v8f acc[4] = {z, z, z, z};

  const int* mrow = mask + (size_t)b * S_ * S_;
  unsigned short* pS = Ps + w * 16 * 40;

  for (int kt = 0; kt < S_; kt += 32) {
    __syncthreads();
#if HAVE_TDM
    if (w == 0) {                                // K tile via Tensor Data Mover
      tdm_load_k_tile(&K[kvBase + (size_t)kt * HD_], (unsigned)(size_t)(void*)Kt);
    }
#else
    {
      int row = tid >> 3, seg = tid & 7;
      cp16(&K[kvBase + (size_t)(kt + row) * HD_ + seg * 8], &Kt[row * 72 + seg * 8]);
    }
#endif
    // V tile: already swizzled in global, contiguous 4KB, async copy
    cp16(vtiles + (size_t)(kt >> 5) * 1024 + tid * 4, &Vt[tid * 4]);
    cp_commit();
#if HAVE_TDM
    if (w == 0) tdm_commit();
#endif
    __syncthreads();

    v8f s0 = z, s1 = z;
    {
      Frag bT = ldsA16(Kt, 72, 0, 0);   s0 = wmma_bf16(aq0, bT, s0);
      Frag bU = ldsA16(Kt, 72, 0, 32);  s0 = wmma_bf16(aq1, bU, s0);
    }
    {
      Frag bT = ldsA16(Kt, 72, 16, 0);  s1 = wmma_bf16(aq0, bT, s1);
      Frag bU = ldsA16(Kt, 72, 16, 32); s1 = wmma_bf16(aq1, bU, s1);
    }

    float v0[8], v1[8], mt[8];
#pragma unroll
    for (int j = 0; j < 8; ++j) {
      int qr = qBase + j + half * 8;
      float a0 = s0[j] * scale + fbias;
      float a1 = s1[j] * scale + fbias;
      if (mrow[(size_t)qr * S_ + kt + r]      == 0) a0 = -1e9f;
      if (mrow[(size_t)qr * S_ + kt + 16 + r] == 0) a1 = -1e9f;
      v0[j] = a0; v1[j] = a1;
      mt[j] = fmaxf(a0, a1);
    }
#pragma unroll
    for (int off = 1; off < 16; off <<= 1)
#pragma unroll
      for (int j = 0; j < 8; ++j) mt[j] = fmaxf(mt[j], __shfl_xor(mt[j], off));

    float rs[8], alpha[8];
#pragma unroll
    for (int j = 0; j < 8; ++j) {
      float mn = fmaxf(m[j], mt[j]);
      alpha[j] = __expf(m[j] - mn);
      m[j] = mn;
      v0[j] = __expf(v0[j] - mn);
      v1[j] = __expf(v1[j] - mn);
      rs[j] = v0[j] + v1[j];
    }
#pragma unroll
    for (int off = 1; off < 16; off <<= 1)
#pragma unroll
      for (int j = 0; j < 8; ++j) rs[j] += __shfl_xor(rs[j], off);
#pragma unroll
    for (int j = 0; j < 8; ++j) l[j] = l[j] * alpha[j] + rs[j];
#pragma unroll
    for (int nt = 0; nt < 4; ++nt)
#pragma unroll
      for (int j = 0; j < 8; ++j) acc[nt][j] *= alpha[j];

#pragma unroll
    for (int j = 0; j < 8; ++j) {                // P -> per-wave LDS scratch
      pS[(j + half * 8) * 40 + r]      = f2bf(v0[j]);
      pS[(j + half * 8) * 40 + 16 + r] = f2bf(v1[j]);
    }
    Frag ap = ldsA16(pS, 40, 0, 0);
#pragma unroll
    for (int nt = 0; nt < 4; ++nt) {
      Frag bv = ldsBsw(Vt, nt * 16);
      acc[nt] = wmma_bf16(ap, bv, acc[nt]);
    }
  }

#pragma unroll
  for (int nt = 0; nt < 4; ++nt)
#pragma unroll
    for (int j = 0; j < 8; ++j) {
      int qr  = qBase + j + half * 8;
      int col = h * HD_ + nt * 16 + r;
      Obuf[((size_t)b * S_ + qr) * D_ + col] = f2bf(acc[nt][j] / l[j]);
    }
}

// ---------------- kernel 6: y = O @ Wo + bo + residual (f32 out) ------------
__global__ __launch_bounds__(256) void k_gemm_out(
    const unsigned short* __restrict__ A, const float* __restrict__ W,
    const float* __restrict__ bias, const float* __restrict__ residual,
    float* __restrict__ Out) {
  __shared__ __attribute__((aligned(16))) unsigned short As[128 * 40];
  __shared__ __attribute__((aligned(16))) unsigned int   Bsw[64 * 16];
  int tid = threadIdx.x;
  int w = tid >> 5, wm = w >> 1, wn = w & 1;
  int lane = tid & 31, half = lane >> 4, r = lane & 15;
  int mBase = blockIdx.x * 128, nBase = blockIdx.y * 64;

  v8f z;
#pragma unroll
  for (int i = 0; i < 8; ++i) z[i] = 0.f;
  v8f c00 = z, c01 = z, c10 = z, c11 = z;

  for (int kb = 0; kb < D_; kb += 32) {
    __syncthreads();
#pragma unroll
    for (int i = 0; i < 2; ++i) {
      int cidx = i * 256 + tid;
      int row = cidx >> 2, seg = cidx & 3;
      cp16(&A[(size_t)(mBase + row) * D_ + kb + seg * 8], &As[row * 40 + seg * 8]);
    }
#pragma unroll
    for (int i = 0; i < 4; ++i) {
      int u = i * 256 + tid;
      int n = u & 63, p = u >> 6;
      float lo = W[(size_t)(kb + 2 * p)     * D_ + nBase + n];
      float hi = W[(size_t)(kb + 2 * p + 1) * D_ + nBase + n];
      Bsw[bswz(n, p)] = (unsigned)f2bf(lo) | ((unsigned)f2bf(hi) << 16);
    }
    cp_commit();
    __syncthreads();

    Frag a0 = ldsA16(As, 40, wm * 32,      0);
    Frag a1 = ldsA16(As, 40, wm * 32 + 16, 0);
    Frag b0 = ldsBsw(Bsw, wn * 32);
    Frag b1 = ldsBsw(Bsw, wn * 32 + 16);
    c00 = wmma_bf16(a0, b0, c00);
    c01 = wmma_bf16(a0, b1, c01);
    c10 = wmma_bf16(a1, b0, c10);
    c11 = wmma_bf16(a1, b1, c11);
  }

  v8f cs[2][2] = {{c00, c01}, {c10, c11}};
#pragma unroll
  for (int sub = 0; sub < 2; ++sub)
#pragma unroll
    for (int nt = 0; nt < 2; ++nt)
#pragma unroll
      for (int j = 0; j < 8; ++j) {
        int mrow = mBase + wm * 32 + sub * 16 + j + half * 8;
        int n = nBase + wn * 32 + nt * 16 + r;
        size_t oi = (size_t)mrow * D_ + n;
        Out[oi] = cs[sub][nt][j] + bias[n] + residual[oi];
      }
}

// ---------------- kernel 7: LayerNorm over rows of 512, in place ------------
__global__ __launch_bounds__(256) void k_ln(float* __restrict__ y,
                                            const float* __restrict__ g,
                                            const float* __restrict__ bta) {
  __shared__ float ss[8], sq[8];
  int row = blockIdx.x, t = threadIdx.x;
  float* p = y + (size_t)row * D_;
  float a = p[t], b2 = p[t + 256];
  float s = a + b2, q = a * a + b2 * b2;
#pragma unroll
  for (int off = 1; off < 32; off <<= 1) {
    s += __shfl_xor(s, off);
    q += __shfl_xor(q, off);
  }
  int w = t >> 5, lane = t & 31;
  if (lane == 0) { ss[w] = s; sq[w] = q; }
  __syncthreads();
  if (t == 0) {
    float S = 0.f, Qs = 0.f;
#pragma unroll
    for (int i = 0; i < 8; ++i) { S += ss[i]; Qs += sq[i]; }
    ss[0] = S; sq[0] = Qs;
  }
  __syncthreads();
  float mean = ss[0] * (1.f / D_);
  float var  = sq[0] * (1.f / D_) - mean * mean;
  float inv  = rsqrtf(var + 1e-5f);
  p[t]       = (a  - mean) * inv * g[t]       + bta[t];
  p[t + 256] = (b2 - mean) * inv * g[t + 256] + bta[t + 256];
}

// ---------------- host launcher ----------------
extern "C" void kernel_launch(void* const* d_in, const int* in_sizes, int n_in,
                              void* d_out, int out_size, void* d_ws, size_t ws_size,
                              hipStream_t stream) {
  (void)in_sizes; (void)n_in; (void)out_size; (void)ws_size;
  const float* x   = (const float*)d_in[0];
  const int*   ids = (const int*)  d_in[1];
  const int*   msk = (const int*)  d_in[2];
  const float* Wq  = (const float*)d_in[3];  const float* bq  = (const float*)d_in[4];
  const float* Wk  = (const float*)d_in[5];  const float* bk  = (const float*)d_in[6];
  const float* Wv  = (const float*)d_in[7];  const float* bv  = (const float*)d_in[8];
  const float* Wo  = (const float*)d_in[9];  const float* bo  = (const float*)d_in[10];
  const float* Wc1 = (const float*)d_in[11]; const float* bc1 = (const float*)d_in[12];
  const float* Wc2 = (const float*)d_in[13]; const float* bc2 = (const float*)d_in[14];
  const float* tf  = (const float*)d_in[15]; const float* rg  = (const float*)d_in[16];
  const float* lng = (const float*)d_in[17]; const float* lnb = (const float*)d_in[18];

  char* ws = (char*)d_ws;
  size_t off = 0;
  auto alloc = [&](size_t bytes) -> char* {
    char* p = ws + off;
    off = (off + bytes + 255) & ~(size_t)255;
    return p;
  };
  unsigned short* xb = (unsigned short*)alloc((size_t)BS_ * D_ * 2);
  unsigned short* qb = (unsigned short*)alloc((size_t)BS_ * D_ * 2);
  unsigned short* kb = (unsigned short*)alloc((size_t)BS_ * D_ * 2);
  unsigned short* vb = (unsigned short*)alloc((size_t)BS_ * D_ * 2);
  unsigned short* ob = (unsigned short*)alloc((size_t)BS_ * D_ * 2);
  float* gc = (float*)alloc((size_t)B_ * D_ * 4);
  float* fb = (float*)alloc((size_t)B_ * H_ * 4);

  k_add_emb<<<(BS_ * D_ / 4) / 256, 256, 0, stream>>>(x, ids, tf, xb);
  k_gc<<<(B_ * D_) / 256, 256, 0, stream>>>(xb, gc);
  k_fb<<<B_, 256, 0, stream>>>(gc, Wc1, bc1, Wc2, bc2, rg, fb);

  dim3 g1(BS_ / 128, D_ / 64);
  k_gemm_proj<1><<<g1, 256, 0, stream>>>(xb, Wq, bq, qb);
  k_gemm_proj<1><<<g1, 256, 0, stream>>>(xb, Wk, bk, kb);
  k_gemm_proj<2><<<g1, 256, 0, stream>>>(xb, Wv, bv, vb);

  dim3 g2(S_ / 128, B_ * H_);
  k_flash<<<g2, 256, 0, stream>>>(qb, kb, vb, msk, fb, ob);

  k_gemm_out<<<g1, 256, 0, stream>>>(ob, Wo, bo, x, (float*)d_out);
  k_ln<<<BS_, 256, 0, stream>>>((float*)d_out, lng, lnb);
}
